// RefinedFeatureAligner_69063074120402
// MI455X (gfx1250) — matrix operand order repaired and checked by reading, not compiled, not run on hardware
//
#include <hip/hip_runtime.h>
#include <hip/hip_bf16.h>

typedef __attribute__((ext_vector_type(16))) __bf16 v16bf;
typedef __attribute__((ext_vector_type(8)))  float  v8f;

constexpr int Hdim = 128, Wdim = 128, HW = Hdim * Wdim;
constexpr int Cch = 64, Tt = 5;
constexpr int CHW   = Cch * HW;        // 1,048,576
constexpr int TCHW  = Tt * CHW;
constexpr int BTCHW = 2 * TCHW;
constexpr int CIN1  = 130;             // 2*C + 2
constexpr int COUT4 = 216;             // 3*K*DG

static __device__ __forceinline__ unsigned short f2bf(float f) {
  unsigned int u = __builtin_bit_cast(unsigned int, f);
  u += 0x7fffu + ((u >> 16) & 1u);     // round-to-nearest-even
  return (unsigned short)(u >> 16);
}
static __device__ __forceinline__ float lrelu(float f) { return f > 0.f ? f : 0.1f * f; }
static __device__ __forceinline__ int iclamp(int v, int lo, int hi) {
  return v < lo ? lo : (v > hi ? hi : v);
}

// ---------------------------------------------------------------------------
// Pack fp32 conv weights [Cout][Cin][3][3] (== GEMM matrix [M][K], K = Cin*9)
// into a WMMA bf16 A-fragment layout with contiguous k order (same order as the B reads):
//   Apack[((mt*KC + kc)*32 + lane)*16 + e], lane<16: M=lane, K=e
//                                           lane>=16: same M, K += 16
// Zero-padded for M >= Cout and K >= Kreal.
// ---------------------------------------------------------------------------
__global__ void pack_w_kernel(const float* __restrict__ w, unsigned short* __restrict__ ap,
                              int COUT, int K, int MT) {
  int KC = (K + 31) >> 5;
  int total = MT * KC * 32 * 16;
  for (int i = blockIdx.x * blockDim.x + threadIdx.x; i < total; i += gridDim.x * blockDim.x) {
    int e    = i & 15;
    int lane = (i >> 4) & 31;
    int blk  = i >> 9;                 // mt*KC + kc
    int kc   = blk % KC, mt = blk / KC;
    int m    = mt * 16 + (lane & 15);
    int kk   = e + ((lane >= 16) ? 16 : 0);
    int kg   = kc * 32 + kk;
    float v  = (m < COUT && kg < K) ? w[m * K + kg] : 0.f;
    ap[i] = f2bf(v);
  }
}

__global__ void fillz_kernel(float* __restrict__ p, int n) {
  int i = blockIdx.x * blockDim.x + threadIdx.x;
  if (i < n) p[i] = 0.f;
}

// ---------------------------------------------------------------------------
// extra = concat(flow_warp(x_i, flow), x_neighbor, flow)   -> bf16 [B][130][H][W]
// ---------------------------------------------------------------------------
__global__ void build_extra_kernel(const float* __restrict__ xi, const float* __restrict__ xnb,
                                   const float* __restrict__ flow, int flow_bstride,
                                   unsigned short* __restrict__ extra) {
  int p = blockIdx.x * blockDim.x + threadIdx.x;
  if (p >= HW) return;
  int bb = blockIdx.y;
  int y = p / Wdim, x = p - y * Wdim;
  float fdx = flow[(size_t)bb * flow_bstride + p];
  float fdy = flow[(size_t)bb * flow_bstride + HW + p];
  float py = (float)y + fdy, px = (float)x + fdx;
  float y0f = floorf(py), x0f = floorf(px);
  int   iy0 = (int)y0f,  ix0 = (int)x0f;
  float wy1 = py - y0f,  wx1 = px - x0f;
  float wy0 = 1.f - wy1, wx0 = 1.f - wx1;
  bool vy0 = (iy0 >= 0) && (iy0 <= Hdim - 1);
  bool vy1 = (iy0 + 1 >= 0) && (iy0 + 1 <= Hdim - 1);
  bool vx0 = (ix0 >= 0) && (ix0 <= Wdim - 1);
  bool vx1 = (ix0 + 1 >= 0) && (ix0 + 1 <= Wdim - 1);
  int cy0 = iclamp(iy0, 0, Hdim - 1), cy1 = iclamp(iy0 + 1, 0, Hdim - 1);
  int cx0 = iclamp(ix0, 0, Wdim - 1), cx1 = iclamp(ix0 + 1, 0, Wdim - 1);
  float wA = wy0 * wx0 * ((vy0 && vx0) ? 1.f : 0.f);
  float wB = wy0 * wx1 * ((vy0 && vx1) ? 1.f : 0.f);
  float wC = wy1 * wx0 * ((vy1 && vx0) ? 1.f : 0.f);
  float wD = wy1 * wx1 * ((vy1 && vx1) ? 1.f : 0.f);
  int a00 = cy0 * Wdim + cx0, a01 = cy0 * Wdim + cx1;
  int a10 = cy1 * Wdim + cx0, a11 = cy1 * Wdim + cx1;
  const float* xb = xi  + (size_t)bb * TCHW;
  const float* nb = xnb + (size_t)bb * TCHW;
  unsigned short* eb = extra + (size_t)bb * CIN1 * HW;
#pragma unroll 4
  for (int c = 0; c < Cch; ++c) {
    const float* ch = xb + (size_t)c * HW;
    float wv = wA * ch[a00] + wB * ch[a01] + wC * ch[a10] + wD * ch[a11];
    eb[(size_t)c * HW + p]        = f2bf(wv);
    eb[(size_t)(64 + c) * HW + p] = f2bf(nb[(size_t)c * HW + p]);
  }
  eb[(size_t)128 * HW + p] = f2bf(fdx);
  eb[(size_t)129 * HW + p] = f2bf(fdy);
}

// ---------------------------------------------------------------------------
// conv3x3 (pad 1) as implicit GEMM with v_wmma_f32_16x16x32_bf16.
// Block = 256 threads = 8 waves; pixel tile = NTILES*16 px of one row.
// Stage im2col [N][KP] (bf16) into LDS, then each wave computes 16x16 tiles.
// MODE 0: out = bf16(lrelu(gemm + bias))
// MODE 1: conv4 fused head -> offbuf (10*tanh + flow_yx) and maskbuf (sigmoid)
// ---------------------------------------------------------------------------
template <int CIN, int MT, int NTILES, int MODE>
__global__ void conv3x3_wmma_kernel(const unsigned short* __restrict__ inp,
                                    const unsigned short* __restrict__ apack,
                                    const float* __restrict__ bias,
                                    unsigned short* __restrict__ out_bf,
                                    float* __restrict__ offbuf,
                                    float* __restrict__ maskbuf,
                                    const float* __restrict__ flow, int flow_bstride) {
  constexpr int K   = CIN * 9;
  constexpr int KC  = (K + 31) / 32;
  constexpr int KP  = KC * 32;
  constexpr int NPX = NTILES * 16;
  constexpr int TPR = Wdim / NPX;
  constexpr int MSTEP = 8 / NTILES;
  extern __shared__ unsigned short s_col[];  // [NPX][KP]

  const int tid = threadIdx.x;
  const int bb  = blockIdx.y;
  const int tileIdx = blockIdx.x;
  const int y   = tileIdx / TPR;
  const int xb0 = (tileIdx % TPR) * NPX;
  const unsigned short* ib = inp + (size_t)bb * CIN * HW;

  // ---- stage im2col tile into LDS (zero pad halo and K padding) ----
  for (int idx = tid; idx < NPX * KP; idx += blockDim.x) {
    int n = idx % NPX, k = idx / NPX;
    unsigned short v = 0;
    if (k < K) {
      int c = k / 9, tap = k - c * 9;
      int ky = tap / 3 - 1, kx = tap - (tap / 3) * 3 - 1;
      int yy = y + ky, xx = xb0 + n + kx;
      if (yy >= 0 && yy < Hdim && xx >= 0 && xx < Wdim)
        v = ib[((size_t)c * Hdim + yy) * Wdim + xx];
    }
    s_col[n * KP + k] = v;
  }
  __syncthreads();

  // ---- WMMA GEMM ----
  const int wave = tid >> 5, lane = tid & 31;
  const int nt   = wave % NTILES, mt0 = wave / NTILES;
  const int ncol = lane & 15, khi = lane >> 4;
  const v16bf* bp    = reinterpret_cast<const v16bf*>(&s_col[(nt * 16 + ncol) * KP + khi * 16]);
  const v16bf* apAll = reinterpret_cast<const v16bf*>(apack);

  for (int mt = mt0; mt < MT; mt += MSTEP) {
    v8f acc = {0.f, 0.f, 0.f, 0.f, 0.f, 0.f, 0.f, 0.f};
    const v16bf* ap = apAll + (size_t)(mt * KC) * 32 + lane;
#pragma unroll 2
    for (int kc = 0; kc < KC; ++kc) {
      if (kc + 4 < KC) __builtin_prefetch(ap + (kc + 4) * 32, 0, 1);
      v16bf a = ap[kc * 32];
      v16bf b = bp[kc * 2];
      acc = __builtin_amdgcn_wmma_f32_16x16x32_bf16(false, a, false, b, (short)0, acc,
                                                    false, false);
    }
    const int px = xb0 + nt * 16 + ncol;
#pragma unroll
    for (int r = 0; r < 8; ++r) {
      int cout = mt * 16 + r + 8 * khi;
      if (MODE == 0) {
        float f = lrelu(acc[r] + bias[cout]);
        out_bf[((size_t)bb * (MT * 16) + cout) * HW + y * Wdim + px] = f2bf(f);
      } else {
        if (cout < COUT4) {
          float f = acc[r] + bias[cout];
          int pix = y * Wdim + px;
          if (cout < 144) {
            int d = cout & 1;  // 0 = dy, 1 = dx; flow_yx[d] = flow[1-d]
            float fl = flow[(size_t)bb * flow_bstride + (size_t)(1 - d) * HW + pix];
            offbuf[((size_t)bb * 144 + cout) * HW + pix] = 10.f * tanhf(f) + fl;
          } else {
            maskbuf[((size_t)bb * 72 + (cout - 144)) * HW + pix] = 1.f / (1.f + expf(-f));
          }
        }
      }
    }
  }
}

// ---------------------------------------------------------------------------
// DCNv2: bilinear sampling (zero pad) * mask -> LDS im2col [32][576] bf16,
// then WMMA GEMM with dcn weights; fp32 store + bias.
// ---------------------------------------------------------------------------
__global__ void dcn_wmma_kernel(const float* __restrict__ xsrc, int x_bstride,
                                const unsigned short* __restrict__ apack,
                                const float* __restrict__ bias,
                                const float* __restrict__ offbuf,
                                const float* __restrict__ maskbuf,
                                float* __restrict__ outp, int out_bstride) {
  constexpr int K = 576, KC = 18, KP = 576, NPX = 32, TPR = Wdim / NPX;
  constexpr int MT = 4, NTILES = 2, MSTEP = 8 / NTILES;
  extern __shared__ unsigned short s_col[];  // [32][576]

  const int tid = threadIdx.x, bb = blockIdx.y;
  const int tileIdx = blockIdx.x;
  const int y   = tileIdx / TPR;
  const int xb0 = (tileIdx % TPR) * NPX;
  const float* xb = xsrc + (size_t)bb * x_bstride;

  // ---- sampling: 32 px * 8 groups * 9 taps positions, 8 channels each ----
  for (int p = tid; p < NPX * 72; p += blockDim.x) {
    int n = p & 31, gt = p >> 5;
    int g = gt / 9, tap = gt - g * 9;
    int x = xb0 + n;
    int pix = y * Wdim + x;
    float offy = offbuf[((size_t)bb * 144 + g * 18 + tap * 2 + 0) * HW + pix];
    float offx = offbuf[((size_t)bb * 144 + g * 18 + tap * 2 + 1) * HW + pix];
    float m    = maskbuf[((size_t)bb * 72 + g * 9 + tap) * HW + pix];
    float py = (float)(y + tap / 3 - 1) + offy;
    float px = (float)(x + (tap - (tap / 3) * 3) - 1) + offx;
    float y0f = floorf(py), x0f = floorf(px);
    int   iy0 = (int)y0f,  ix0 = (int)x0f;
    float wy1 = py - y0f,  wx1 = px - x0f;
    float wy0 = 1.f - wy1, wx0 = 1.f - wx1;
    bool vy0 = (iy0 >= 0) && (iy0 <= Hdim - 1);
    bool vy1 = (iy0 + 1 >= 0) && (iy0 + 1 <= Hdim - 1);
    bool vx0 = (ix0 >= 0) && (ix0 <= Wdim - 1);
    bool vx1 = (ix0 + 1 >= 0) && (ix0 + 1 <= Wdim - 1);
    int cy0 = iclamp(iy0, 0, Hdim - 1), cy1 = iclamp(iy0 + 1, 0, Hdim - 1);
    int cx0 = iclamp(ix0, 0, Wdim - 1), cx1 = iclamp(ix0 + 1, 0, Wdim - 1);
    float wA = wy0 * wx0 * ((vy0 && vx0) ? 1.f : 0.f);
    float wB = wy0 * wx1 * ((vy0 && vx1) ? 1.f : 0.f);
    float wC = wy1 * wx0 * ((vy1 && vx0) ? 1.f : 0.f);
    float wD = wy1 * wx1 * ((vy1 && vx1) ? 1.f : 0.f);
    int a00 = cy0 * Wdim + cx0, a01 = cy0 * Wdim + cx1;
    int a10 = cy1 * Wdim + cx0, a11 = cy1 * Wdim + cx1;
    const float* chp = xb + (size_t)(g * 8) * HW;
#pragma unroll
    for (int c = 0; c < 8; ++c) {
      const float* ch = chp + (size_t)c * HW;
      float val = (wA * ch[a00] + wB * ch[a01] + wC * ch[a10] + wD * ch[a11]) * m;
      s_col[n * KP + (g * 8 + c) * 9 + tap] = f2bf(val);
    }
  }
  __syncthreads();

  // ---- WMMA GEMM: 64 Cout x 32 px, K = 576 ----
  const int wave = tid >> 5, lane = tid & 31;
  const int nt   = wave % NTILES, mt0 = wave / NTILES;
  const int ncol = lane & 15, khi = lane >> 4;
  const v16bf* bp    = reinterpret_cast<const v16bf*>(&s_col[(nt * 16 + ncol) * KP + khi * 16]);
  const v16bf* apAll = reinterpret_cast<const v16bf*>(apack);

  for (int mt = mt0; mt < MT; mt += MSTEP) {
    v8f acc = {0.f, 0.f, 0.f, 0.f, 0.f, 0.f, 0.f, 0.f};
    const v16bf* ap = apAll + (size_t)(mt * KC) * 32 + lane;
#pragma unroll 2
    for (int kc = 0; kc < KC; ++kc) {
      v16bf a = ap[kc * 32];
      v16bf b = bp[kc * 2];
      acc = __builtin_amdgcn_wmma_f32_16x16x32_bf16(false, a, false, b, (short)0, acc,
                                                    false, false);
    }
    const int px = xb0 + nt * 16 + ncol;
#pragma unroll
    for (int r = 0; r < 8; ++r) {
      int cout = mt * 16 + r + 8 * khi;
      outp[(size_t)bb * out_bstride + (size_t)cout * HW + y * Wdim + px] = acc[r] + bias[cout];
    }
  }
}

// ---------------------------------------------------------------------------
// Host orchestration
// ---------------------------------------------------------------------------
extern "C" void kernel_launch(void* const* d_in, const int* in_sizes, int n_in,
                              void* d_out, int out_size, void* d_ws, size_t ws_size,
                              hipStream_t stream) {
  (void)in_sizes; (void)n_in; (void)out_size; (void)ws_size;
  const float* x  = (const float*)d_in[0];
  const float* fb = (const float*)d_in[1];
  const float* ff = (const float*)d_in[2];
  const float* w1 = (const float*)d_in[3];  const float* b1 = (const float*)d_in[4];
  const float* w2 = (const float*)d_in[5];  const float* b2 = (const float*)d_in[6];
  const float* w3 = (const float*)d_in[7];  const float* b3 = (const float*)d_in[8];
  const float* w4 = (const float*)d_in[9];  const float* b4 = (const float*)d_in[10];
  const float* dw = (const float*)d_in[11]; const float* db = (const float*)d_in[12];
  float* out = (float*)d_out;

  char* wsp = (char*)d_ws;
  auto alloc = [&](size_t bytes) -> char* {
    char* p = wsp;
    wsp += (bytes + 255) & ~size_t(255);
    return p;
  };
  unsigned short* extra = (unsigned short*)alloc((size_t)2 * CIN1 * HW * 2);
  unsigned short* h1    = (unsigned short*)alloc((size_t)2 * 64 * HW * 2);
  unsigned short* h2    = (unsigned short*)alloc((size_t)2 * 64 * HW * 2);
  float* offb  = (float*)alloc((size_t)2 * 144 * HW * 4);
  float* maskb = (float*)alloc((size_t)2 * 72 * HW * 4);
  const int KC1 = (CIN1 * 9 + 31) / 32;  // 37
  const int KC2 = 18;
  unsigned short* A1 = (unsigned short*)alloc((size_t)4 * KC1 * 512 * 2);
  unsigned short* A2 = (unsigned short*)alloc((size_t)4 * KC2 * 512 * 2);
  unsigned short* A3 = (unsigned short*)alloc((size_t)4 * KC2 * 512 * 2);
  unsigned short* A4 = (unsigned short*)alloc((size_t)14 * KC2 * 512 * 2);
  unsigned short* AD = (unsigned short*)alloc((size_t)4 * KC2 * 512 * 2);

  // Pack weights (once per launch; deterministic)
  auto packLaunch = [&](const float* w, unsigned short* ap, int COUT, int K, int MT) {
    int total = MT * ((K + 31) / 32) * 512;
    pack_w_kernel<<<dim3((total + 255) / 256), dim3(256), 0, stream>>>(w, ap, COUT, K, MT);
  };
  packLaunch(w1, A1, 64, CIN1 * 9, 4);
  packLaunch(w2, A2, 64, 576, 4);
  packLaunch(w3, A3, 64, 576, 4);
  packLaunch(w4, A4, COUT4, 576, 14);
  packLaunch(dw, AD, 64, 576, 4);

  // Zero the untouched output slices: backward t=4, forward t=0 (both b)
  {
    int n = CHW, grid = (n + 255) / 256;
    fillz_kernel<<<dim3(grid), dim3(256), 0, stream>>>(out + (size_t)(0 * Tt + 4) * CHW, n);
    fillz_kernel<<<dim3(grid), dim3(256), 0, stream>>>(out + (size_t)(1 * Tt + 4) * CHW, n);
    fillz_kernel<<<dim3(grid), dim3(256), 0, stream>>>(out + (size_t)BTCHW + (size_t)(0 * Tt) * CHW, n);
    fillz_kernel<<<dim3(grid), dim3(256), 0, stream>>>(out + (size_t)BTCHW + (size_t)(1 * Tt) * CHW, n);
  }

  const int FBSTR = (Tt - 1) * 2 * HW;     // per-b stride of flows tensors
  const dim3 blk(256);
  const unsigned smem1 = 16 * KC1 * 32 * 2;   // 37,888 B (conv1, 16-px tile)
  const unsigned smem2 = 32 * 576 * 2;        // 36,864 B

  auto run_gdcn = [&](const float* xi, const float* xnb, const float* flow, float* osl) {
    build_extra_kernel<<<dim3(HW / 256, 2), blk, 0, stream>>>(xi, xnb, flow, FBSTR, extra);
    conv3x3_wmma_kernel<CIN1, 4, 1, 0><<<dim3(Hdim * (Wdim / 16), 2), blk, smem1, stream>>>(
        extra, A1, b1, h1, nullptr, nullptr, nullptr, 0);
    conv3x3_wmma_kernel<64, 4, 2, 0><<<dim3(Hdim * (Wdim / 32), 2), blk, smem2, stream>>>(
        h1, A2, b2, h2, nullptr, nullptr, nullptr, 0);
    conv3x3_wmma_kernel<64, 4, 2, 0><<<dim3(Hdim * (Wdim / 32), 2), blk, smem2, stream>>>(
        h2, A3, b3, h1, nullptr, nullptr, nullptr, 0);
    conv3x3_wmma_kernel<64, 14, 2, 1><<<dim3(Hdim * (Wdim / 32), 2), blk, smem2, stream>>>(
        h1, A4, b4, nullptr, offb, maskb, flow, FBSTR);
    dcn_wmma_kernel<<<dim3(Hdim * (Wdim / 32), 2), blk, smem2, stream>>>(
        xi, TCHW, AD, db, offb, maskb, osl, TCHW);
  };

  // Backward: output slice t (0..3) uses i = t+1, neighbor x[:,t], flow fb[:,t]
  for (int t = 0; t < 4; ++t) {
    run_gdcn(x + (size_t)(t + 1) * CHW, x + (size_t)t * CHW,
             fb + (size_t)t * 2 * HW, out + (size_t)t * CHW);
  }
  // Forward: output slice t (1..4) uses i = t-1, neighbor x[:,t], flow ff[:,t-1]
  for (int t = 1; t <= 4; ++t) {
    run_gdcn(x + (size_t)(t - 1) * CHW, x + (size_t)t * CHW,
             ff + (size_t)(t - 1) * 2 * HW, out + (size_t)BTCHW + (size_t)t * CHW);
  }
}
